// NNConvNet_1layer_88553635709218
// MI455X (gfx1250) — matrix operand order repaired and checked by reading, not compiled, not run on hardware
//
#include <hip/hip_runtime.h>
#include <hip/hip_bf16.h>
#include <math.h>

typedef __attribute__((ext_vector_type(16))) _Float16 v16h;
typedef __attribute__((ext_vector_type(8)))  float    v8f;

constexpr int kN    = 50000;
constexpr int kE    = 800000;
constexpr int kIN   = 16;
constexpr int kOUT  = 32;
constexpr int kHID  = 25;
constexpr int kC    = 10;
constexpr int kTiles = kE / 16;        // 50000 16-edge tiles
constexpr int kEdgeBlocks = 625;       // 625*8 = 5000 waves
constexpr int kWaveStride = kEdgeBlocks * 8;      // 5000
constexpr int kIters = kTiles / (2 * kWaveStride); // 5 iters x 2 tiles = 10 tiles/wave

// Non-returning device-scope f32 max atomic (GLOBAL_ATOMIC_MAX_NUM_F32).
__device__ __forceinline__ void atomic_max_f32(float* addr, float val) {
  asm volatile("global_atomic_max_num_f32 %0, %1, off scope:SCOPE_DEV"
               :: "v"(addr), "v"(val) : "memory");
}

__global__ void init_agg_kernel(float* __restrict__ agg, int n) {
  int i = blockIdx.x * blockDim.x + threadIdx.x;
  if (i < n) agg[i] = -__builtin_inff();
}

// One wave per pair of 16-edge tiles (B fragments reused across the pair).
// msg[16e x 32o] = hext[16e x 26] @ w2ext[26 x 512]  (WMMA, K padded to 32),
// then per-edge contraction with x[src] on VALU (co-executes with WMMA),
// then scatter-max via f32 atomics.
__launch_bounds__(256, 2)
__global__ void edge_kernel(const float* __restrict__ x,
                            const float* __restrict__ edge_attr,
                            const int*   __restrict__ edge_index,
                            const float* __restrict__ w1,
                            const float* __restrict__ b1,
                            const float* __restrict__ w2,
                            const float* __restrict__ b2,
                            float*       __restrict__ agg) {
  // B fragments of w2ext in WMMA layout: [chunk][lane][16 halves], 32 KB.
  //   lane<16:  N = lane,    halves j<8 -> K = j,   j>=8 -> K = 16+(j-8)
  //   lane>=16: N = lane-16, halves j<8 -> K = 8+j, j>=8 -> K = 24+(j-8)
  __shared__ _Float16 w2f[32][32][16];
  __shared__ float    xs[8][2][16][16];  // per-wave gathered x[src], 2 tiles
  __shared__ int      sdst[8][2][16];    // per-wave dst indices, 2 tiles
  __shared__ float    sw1[32], sb1[32];

  const int tid  = threadIdx.x;
  const int lane = tid & 31;
  const int wave = tid >> 5;

  // Stage padded edge-MLP layer-1: k<25 real; k==25 -> relu(0*a+1)=1 (bias row); k>25 -> 0.
  if (tid < 32) {
    float wv = 0.0f, bv = 0.0f;
    if (tid < kHID)       { wv = w1[tid]; bv = b1[tid]; }
    else if (tid == kHID) { bv = 1.0f; }
    sw1[tid] = wv; sb1[tid] = bv;
  }
  // Stage w2ext into B-fragment layout (f16). Row K==25 carries b2 (bias fold).
  for (int idx = tid; idx < 32 * 32 * 16; idx += 256) {
    int j = idx & 15;
    int l = (idx >> 4) & 31;
    int c = idx >> 9;
    int col = c * 16 + (l & 15);
    int K   = ((j & 8) ? 16 : 0) + ((l < 16) ? 0 : 8) + (j & 7);
    float v = 0.0f;
    if (K < kHID)       v = w2[K * 512 + col];
    else if (K == kHID) v = b2[col];
    w2f[c][l][j] = (_Float16)v;
  }
  __syncthreads();

  const int* srcIdx = edge_index;        // edge_index[0] = sources
  const int* dstIdx = edge_index + kE;   // edge_index[1] = targets
  const int row   = lane & 15;           // A-matrix M row = edge-in-tile
  const int kb    = (lane < 16) ? 0 : 8; // A-fragment K base per lane half
  const int erow0 = (lane < 16) ? 0 : 8; // D-tile M rows this lane holds

  const int waveGlobal = blockIdx.x * 8 + wave;

  for (int it = 0; it < kIters; ++it) {
    const int tileA = waveGlobal + (2 * it)     * kWaveStride;
    const int tileB = waveGlobal + (2 * it + 1) * kWaveStride;
    const int eA0 = tileA * 16;
    const int eB0 = tileB * 16;

    // Gather x[src] rows (2 tiles x 16 edges x 16 feats) + dst indices into LDS.
    {
      int e = lane >> 1, half = lane & 1;
      int sA = srcIdx[eA0 + e];
      int sB = srcIdx[eB0 + e];
      const float4* xrA = (const float4*)(x + (size_t)sA * kIN);
      const float4* xrB = (const float4*)(x + (size_t)sB * kIN);
      float4* xwA = (float4*)&xs[wave][0][e][0];
      float4* xwB = (float4*)&xs[wave][1][e][0];
      xwA[half * 2 + 0] = xrA[half * 2 + 0];
      xwA[half * 2 + 1] = xrA[half * 2 + 1];
      xwB[half * 2 + 0] = xrB[half * 2 + 0];
      xwB[half * 2 + 1] = xrB[half * 2 + 1];
      int tsel = (lane < 16) ? 0 : 1;
      sdst[wave][tsel][row] = dstIdx[(tsel ? eB0 : eA0) + row];
    }
    __syncthreads();

    // Build A fragments: h rows for this lane's edge in each tile (from scalar a_e).
    float aA = edge_attr[eA0 + row];
    float aB = edge_attr[eB0 + row];
    v16h afragA, afragB;
#pragma unroll
    for (int j = 0; j < 8; ++j) {
      int k1 = kb + j;
      int k2 = 16 + kb + j;
      float w1k1 = sw1[k1], b1k1 = sb1[k1];
      float w1k2 = sw1[k2], b1k2 = sb1[k2];
      afragA[j]     = (_Float16)fmaxf(fmaf(aA, w1k1, b1k1), 0.0f);
      afragA[8 + j] = (_Float16)fmaxf(fmaf(aA, w1k2, b1k2), 0.0f);
      afragB[j]     = (_Float16)fmaxf(fmaf(aB, w1k1, b1k1), 0.0f);
      afragB[8 + j] = (_Float16)fmaxf(fmaf(aB, w1k2, b1k2), 0.0f);
    }

    float accA0[8], accA1[8], accB0[8], accB1[8];
#pragma unroll
    for (int r = 0; r < 8; ++r) {
      accA0[r] = 0.0f; accA1[r] = 0.0f;
      accB0[r] = 0.0f; accB1[r] = 0.0f;
    }

    // 32 column-chunks of W_e; chunk c covers cols [16c,16c+16) -> i=c>>1, o=row+16*(c&1).
    // Each B fragment feeds two independent WMMAs (tile A / tile B) so their
    // consumes fill each other's WMMA->VALU hazard slots.
    for (int t = 0; t < 16; ++t) {       // t == input feature index i
      float xvA[8], xvB[8];
#pragma unroll
      for (int r = 0; r < 8; ++r) {
        xvA[r] = xs[wave][0][erow0 + r][t];
        xvB[r] = xs[wave][1][erow0 + r][t];
      }
#pragma unroll
      for (int par = 0; par < 2; ++par) {
        int c = 2 * t + par;
        v16h bfrag = *(const v16h*)(&w2f[c][lane][0]);
        v8f  czA = {};
        v8f  czB = {};
        v8f  dA = __builtin_amdgcn_wmma_f32_16x16x32_f16(
                    false, afragA, false, bfrag, (short)0, czA, false, false);
        v8f  dB = __builtin_amdgcn_wmma_f32_16x16x32_f16(
                    false, afragB, false, bfrag, (short)0, czB, false, false);
        float* aAcc = par ? accA1 : accA0;
        float* aBcc = par ? accB1 : accB0;
#pragma unroll
        for (int r = 0; r < 8; ++r) aAcc[r] = fmaf(xvA[r], dA[r], aAcc[r]);
#pragma unroll
        for (int r = 0; r < 8; ++r) aBcc[r] = fmaf(xvB[r], dB[r], aBcc[r]);
      }
    }

    // Scatter-max: per tile, lane holds msg for 8 edges x 2 output cols.
#pragma unroll
    for (int r = 0; r < 8; ++r) {
      int dnA = sdst[wave][0][erow0 + r];
      int dnB = sdst[wave][1][erow0 + r];
      float* baseA = agg + (size_t)dnA * kOUT + row;
      float* baseB = agg + (size_t)dnB * kOUT + row;
      atomic_max_f32(baseA,      accA0[r]);
      atomic_max_f32(baseA + 16, accA1[r]);
      atomic_max_f32(baseB,      accB0[r]);
      atomic_max_f32(baseB + 16, accB1[r]);
    }
    __syncthreads();
  }
}

__global__ void node_kernel(const float* __restrict__ x,
                            const float* __restrict__ agg,
                            const float* __restrict__ root,
                            const float* __restrict__ bias,
                            const float* __restrict__ fcw,
                            const float* __restrict__ fcb,
                            float* __restrict__ out) {
  __shared__ float sroot[kIN * kOUT];
  __shared__ float sbias[kOUT];
  __shared__ float sfw[kOUT * kC];
  __shared__ float sfb[kC];
  for (int i = threadIdx.x; i < kIN * kOUT; i += blockDim.x) sroot[i] = root[i];
  for (int i = threadIdx.x; i < kOUT;       i += blockDim.x) sbias[i] = bias[i];
  for (int i = threadIdx.x; i < kOUT * kC;  i += blockDim.x) sfw[i]  = fcw[i];
  for (int i = threadIdx.x; i < kC;         i += blockDim.x) sfb[i]  = fcb[i];
  __syncthreads();

  int n = blockIdx.x * blockDim.x + threadIdx.x;
  if (n >= kN) return;

  float xi[kIN];
#pragma unroll
  for (int i = 0; i < kIN; ++i) xi[i] = x[(size_t)n * kIN + i];

  float o[kOUT];
#pragma unroll
  for (int j = 0; j < kOUT; ++j) {
    float a = agg[(size_t)n * kOUT + j];
    unsigned bits = __float_as_uint(a);
    if ((bits & 0x7F800000u) == 0x7F800000u) a = 0.0f;  // -inf (empty) / non-finite -> 0
    float t = sbias[j];
#pragma unroll
    for (int i = 0; i < kIN; ++i) t = fmaf(xi[i], sroot[i * kOUT + j], t);
    float z = a + t;
    o[j] = (z > 0.0f) ? z : (expf(z) - 1.0f);           // ELU
  }

  float lg[kC];
  float m = -__builtin_inff();
#pragma unroll
  for (int c = 0; c < kC; ++c) {
    float t = sfb[c];
#pragma unroll
    for (int j = 0; j < kOUT; ++j) t = fmaf(o[j], sfw[j * kC + c], t);
    lg[c] = t;
    m = fmaxf(m, t);
  }
  float s = 0.0f;
#pragma unroll
  for (int c = 0; c < kC; ++c) s += expf(lg[c] - m);
  float ls = logf(s);
#pragma unroll
  for (int c = 0; c < kC; ++c) out[(size_t)n * kC + c] = lg[c] - m - ls;
}

extern "C" void kernel_launch(void* const* d_in, const int* in_sizes, int n_in,
                              void* d_out, int out_size, void* d_ws, size_t ws_size,
                              hipStream_t stream) {
  const float* x         = (const float*)d_in[0];
  const float* edge_attr = (const float*)d_in[1];
  const int*   edge_idx  = (const int*)  d_in[2];   // JAX (x64 off) => int32 [2,E]
  const float* w1        = (const float*)d_in[3];
  const float* b1        = (const float*)d_in[4];
  const float* w2        = (const float*)d_in[5];
  const float* b2        = (const float*)d_in[6];
  const float* root      = (const float*)d_in[7];
  const float* bias      = (const float*)d_in[8];
  const float* fcw       = (const float*)d_in[9];
  const float* fcb       = (const float*)d_in[10];
  float* out = (float*)d_out;
  float* agg = (float*)d_ws;                        // kN*kOUT f32 = 6.4 MB scratch

  init_agg_kernel<<<(kN * kOUT + 255) / 256, 256, 0, stream>>>(agg, kN * kOUT);
  edge_kernel<<<kEdgeBlocks, 256, 0, stream>>>(x, edge_attr, edge_idx,
                                               w1, b1, w2, b2, agg);
  node_kernel<<<(kN + 255) / 256, 256, 0, stream>>>(x, agg, root, bias, fcw, fcb, out);
}
